// MultiHeadAttention_21406117003431
// MI455X (gfx1250) — matrix operand order repaired and checked by reading, not compiled, not run on hardware
//
#include <hip/hip_runtime.h>
#include <hip/hip_fp16.h>

// ---------------------------------------------------------------------------
// MHA forward for MI455X (gfx1250): f16 WMMA everywhere, f32 softmax/accum.
// GEMM tiles are fed by the Tensor Data Mover (tensor_load_to_lds) with
// double-buffered LDS; attention streams K and V^T as contiguous B-fragments.
// ---------------------------------------------------------------------------

typedef __attribute__((ext_vector_type(16))) _Float16     v16h;
typedef __attribute__((ext_vector_type(8)))  _Float16     v8h;
typedef __attribute__((ext_vector_type(4)))  _Float16     v4h;
typedef __attribute__((ext_vector_type(8)))  float        v8f;
typedef __attribute__((ext_vector_type(4)))  unsigned int v4u;
typedef __attribute__((ext_vector_type(8)))  int          v8i;
typedef __attribute__((ext_vector_type(4)))  int          v4i;

#define D_MODEL 1024
#define N_HEADS 16
#define D_K     64
#define BATCH   4
#define SEQ     2048
#define MTOT    (BATCH * SEQ)          // 8192 rows

// ---------------- cross-lane 16-wide row reductions via ds_swizzle ----------
template <int PAT>
__device__ __forceinline__ float swz(float x) {
    return __int_as_float(__builtin_amdgcn_ds_swizzle(__float_as_int(x), PAT));
}
__device__ __forceinline__ float rowmax16(float x) {
    x = fmaxf(x, swz<0x041f>(x));
    x = fmaxf(x, swz<0x081f>(x));
    x = fmaxf(x, swz<0x101f>(x));
    x = fmaxf(x, swz<0x201f>(x));
    return x;
}
__device__ __forceinline__ float rowsum16(float x) {
    x += swz<0x041f>(x);
    x += swz<0x081f>(x);
    x += swz<0x101f>(x);
    x += swz<0x201f>(x);
    return x;
}

// ---------------- WMMA fragment loaders (ISA 7.12.2 layouts) ----------------
// A (16x32 f16, MxK), row-major source, leading dim ld.
__device__ __forceinline__ v16h load_a_frag(const _Float16* __restrict__ base, int ld) {
    const int lane = threadIdx.x & 31;
    const int row  = lane & 15;
    const int kb   = (lane >> 4) << 3;           // 0 or 8
    const _Float16* p = base + row * ld + kb;
    v8h lo = *(const v8h*)(p);
    v8h hi = *(const v8h*)(p + 16);
    v16h a;
#pragma unroll
    for (int i = 0; i < 8; ++i) { a[i] = lo[i]; a[i + 8] = hi[i]; }
    return a;
}

// B (32x16 f16, KxN) where memory holds B^T row-major: src[n][k], ld elements.
__device__ __forceinline__ v16h load_b_frag(const _Float16* __restrict__ base, int ld) {
    const int lane = threadIdx.x & 31;
    const int n = lane & 15;
    const int g = lane >> 4;
    const _Float16* p = base + n * ld + g * 8;
    v8h lo = *(const v8h*)(p);
    v8h hi = *(const v8h*)(p + 16);
    v16h b;
#pragma unroll
    for (int i = 0; i < 8; ++i) { b[i] = lo[i]; b[i + 8] = hi[i]; }
    return b;
}

__device__ __forceinline__ v8f wmma_f16(v16h a, v16h b, v8f c) {
    return __builtin_amdgcn_wmma_f32_16x16x32_f16(false, a, false, b, (short)0, c,
                                                  false, false);
}

// ---------------- Tensor Data Mover descriptors (ISA 8.3/8.4) ---------------
// group0 = {count=1, lds_addr, global_addr[31:0], global_addr[56:32]|type=2}
// group1 = {data_size=2B, tensor dims, tile dims, dim0 stride}
__device__ __forceinline__ v4u tdm_group0(unsigned lds_addr, const _Float16* gptr) {
    const unsigned long long ga = (unsigned long long)(uintptr_t)gptr;
    v4u g0;
    g0[0] = 1u;                                                // count=1, user mode
    g0[1] = lds_addr;                                          // LDS byte address
    g0[2] = (unsigned)ga;                                      // addr [31:0]
    g0[3] = (unsigned)((ga >> 32) & 0x01ffffffu) | (2u << 30); // [56:32] | type=2
    return g0;
}
__device__ __forceinline__ v8i tdm_group1(unsigned tile_d0, unsigned tile_d1,
                                          unsigned tensor_d0, unsigned tensor_d1,
                                          unsigned stride0) {
    v8i g1;
    g1[0] = (int)(1u << 16);                               // data_size=1 -> 2 bytes
    g1[1] = (int)((tensor_d0 & 0xffffu) << 16);            // tensor_dim0[15:0]
    g1[2] = (int)((tensor_d0 >> 16) | ((tensor_d1 & 0xffffu) << 16));
    g1[3] = (int)((tensor_d1 >> 16) | (tile_d0 << 16));    // | tile_dim0
    g1[4] = (int)(tile_d1 & 0xffffu);                      // tile_dim1 (tile_dim2=0)
    g1[5] = (int)stride0;                                  // tensor_dim0_stride[31:0]
    g1[6] = 0;
    g1[7] = 0;
    return g1;
}
__device__ __forceinline__ void tdm_issue(v4u g0, v8i g1) {
    v4i g2 = {0, 0, 0, 0};
    v4i g3 = {0, 0, 0, 0};
#if __clang_major__ >= 23
    v8i g4 = {0, 0, 0, 0, 0, 0, 0, 0};
    __builtin_amdgcn_tensor_load_to_lds(g0, g1, g2, g3, g4, 0);
#else
    __builtin_amdgcn_tensor_load_to_lds(g0, g1, g2, g3, 0);
#endif
}

// ---------------- f32 -> f16 conversion (vectorized) ------------------------
__global__ __launch_bounds__(256) void cvt_f32_to_f16_kernel(
    const float* __restrict__ in, _Float16* __restrict__ out, int n4) {
    int i = blockIdx.x * 256 + threadIdx.x;
    if (i < n4) {
        float4 f = ((const float4*)in)[i];
        v4h o;
        o[0] = (_Float16)f.x; o[1] = (_Float16)f.y;
        o[2] = (_Float16)f.z; o[3] = (_Float16)f.w;
        ((v4h*)out)[i] = o;
    }
}

// ---------------- tiled GEMM: Y = X(MxK) * W(NxK)^T -------------------------
// 256 threads = 8 waves; WG tile 128(M) x 64(N); wave tile 32x32; K step 32.
// Tiles are DMA'd by the TDM into double-buffered LDS; the D# is built once
// and only the LDS-buffer toggle / global K-advance mutate per iteration.
// Modes: 0 = f16 (B,H,S,64); 1 = f32 row-major; 2 = f16 transposed (B,H,64,S).
__global__ __launch_bounds__(256) void gemm_xwT_kernel(
    const _Float16* __restrict__ X, const _Float16* __restrict__ W,
    _Float16* __restrict__ Yh, float* __restrict__ Yf, int mode) {
    __shared__ __align__(128) _Float16 xs[2][128 * 32];    // 8 KB per buffer
    __shared__ __align__(128) _Float16 wsm[2][64 * 32];    // 4 KB per buffer

    const int tid  = threadIdx.x;
    const int wid  = tid >> 5;
    const int lane = tid & 31;
    const int m0 = blockIdx.y * 128;
    const int n0 = blockIdx.x * 64;
    const int wm = (wid & 3) * 32;
    const int wn = (wid >> 2) * 32;

    v8f c[2][2] = {};

    // Loop-invariant D# groups; address words mutated in-loop.
    v4u g0x = tdm_group0((unsigned)(uintptr_t)&xs[0][0], X + (size_t)m0 * D_MODEL);
    v8i g1x = tdm_group1(32, 128, D_MODEL, MTOT, D_MODEL);
    v4u g0w = tdm_group0((unsigned)(uintptr_t)&wsm[0][0], W + (size_t)n0 * D_MODEL);
    v8i g1w = tdm_group1(32, 64, D_MODEL, D_MODEL, D_MODEL);

    // Prologue: DMA first tile pair into buffer 0.
    if (wid == 0) {
        tdm_issue(g0x, g1x);
        tdm_issue(g0w, g1w);
    }

    int buf = 0;
    for (int k0 = 0; k0 < D_MODEL; k0 += 32, buf ^= 1) {
        if (wid == 0) {
            if (k0 + 32 < D_MODEL) {
                // Advance D# to the next K tile / other LDS buffer, issue, then
                // wait for the current pair (<=2 outstanding = the two just issued).
                g0x[1] ^= (unsigned)(128 * 32 * 2);   // LDS double-buffer toggle
                g0x[2] += 64u;                        // K += 32 halfs = 64 bytes
                g0w[1] ^= (unsigned)(64 * 32 * 2);
                g0w[2] += 64u;
                tdm_issue(g0x, g1x);
                tdm_issue(g0w, g1w);
                __builtin_amdgcn_s_wait_tensorcnt(2);
            } else {
                __builtin_amdgcn_s_wait_tensorcnt(0);
            }
        }
        __syncthreads();

        v16h a0 = load_a_frag(&xs[buf][(wm + 0) * 32], 32);
        v16h a1 = load_a_frag(&xs[buf][(wm + 16) * 32], 32);
        v16h b0 = load_b_frag(&wsm[buf][(wn + 0) * 32], 32);
        v16h b1 = load_b_frag(&wsm[buf][(wn + 16) * 32], 32);

        c[0][0] = wmma_f16(a0, b0, c[0][0]);
        c[0][1] = wmma_f16(a0, b1, c[0][1]);
        c[1][0] = wmma_f16(a1, b0, c[1][0]);
        c[1][1] = wmma_f16(a1, b1, c[1][1]);

        __syncthreads();
    }

    // Epilogue: C/D layout -> row = r + 8*(lane>=16), col = lane&15
    const int g = lane >> 4;
    const int nl = lane & 15;
#pragma unroll
    for (int i = 0; i < 2; ++i)
#pragma unroll
        for (int j = 0; j < 2; ++j)
#pragma unroll
            for (int r = 0; r < 8; ++r) {
                const int m = m0 + wm + i * 16 + r + g * 8;
                const int n = n0 + wn + j * 16 + nl;
                const float val = c[i][j][r];
                if (mode == 0) {
                    const int b = m >> 11, s = m & (SEQ - 1);
                    const int h = n >> 6, d = n & (D_K - 1);
                    Yh[(((size_t)(b * N_HEADS + h) * SEQ) + s) * D_K + d] =
                        (_Float16)val;
                } else if (mode == 2) {
                    const int b = m >> 11, s = m & (SEQ - 1);
                    const int h = n >> 6, d = n & (D_K - 1);
                    Yh[(((size_t)(b * N_HEADS + h) * D_K) + d) * SEQ + s] =
                        (_Float16)val;
                } else {
                    Yf[(size_t)m * D_MODEL + n] = val;
                }
            }
}

// ---------------- flash attention (causal, online softmax) ------------------
// Grid: (SEQ/128, BATCH*N_HEADS), 256 threads (8 waves). Each wave owns 16
// query rows; streams 32-key blocks with early causal exit; no cross-wave
// barriers in the loop. V is consumed pre-transposed (B,H,64,S), so both K
// and V operands are row-contiguous B-fragment loads (two 16B v8h per lane).
__global__ __launch_bounds__(256) void attn_kernel(
    const _Float16* __restrict__ Q, const _Float16* __restrict__ K,
    const _Float16* __restrict__ Vt, _Float16* __restrict__ O) {
    __shared__ __align__(32) _Float16 plds[8 * 16 * 32];   // per-wave 16x32 P tile

    const int tid  = threadIdx.x;
    const int wid  = tid >> 5;
    const int lane = tid & 31;
    const int g  = lane >> 4;
    const int nl = lane & 15;

    const int bh = blockIdx.y;                 // b*N_HEADS + h
    const int b = bh >> 4, h = bh & (N_HEADS - 1);
    const int qbase = blockIdx.x * 128 + wid * 16;

    const _Float16* Qb  = Q  + (size_t)bh * SEQ * D_K;
    const _Float16* Kb  = K  + (size_t)bh * SEQ * D_K;
    const _Float16* Vtb = Vt + (size_t)bh * D_K * SEQ;     // [d][s]
    _Float16* pw = plds + wid * (16 * 32);

    // Q fragments for all 16 rows x 64 d (two K=32 halves), loaded once.
    const v16h aq0 = load_a_frag(Qb + (size_t)qbase * D_K, D_K);
    const v16h aq1 = load_a_frag(Qb + (size_t)qbase * D_K + 32, D_K);

    v8f acc[4] = {};
    float mrow[8], lrow[8];
#pragma unroll
    for (int r = 0; r < 8; ++r) { mrow[r] = -INFINITY; lrow[r] = 0.f; }

    const float scale = 0.125f;                // 1/sqrt(64)
    const int qhi = qbase + 15;

    for (int j0 = 0; j0 <= qhi; j0 += 32) {
        // speculative prefetch of the next key block (safe past the end)
        __builtin_prefetch(Kb + (size_t)(j0 + 32) * D_K, 0, 0);
        __builtin_prefetch(Vtb + (j0 + 32), 0, 0);

        // ---- scores: 16 queries x 32 keys, contraction d=64 (2 WMMA each) --
        v8f s[2];
#pragma unroll
        for (int kn = 0; kn < 2; ++kn) {
            v16h bk0 = load_b_frag(Kb + (size_t)(j0 + kn * 16) * D_K, D_K);
            v16h bk1 = load_b_frag(Kb + (size_t)(j0 + kn * 16) * D_K + 32, D_K);
            v8f z = {};
            z = wmma_f16(aq0, bk0, z);
            z = wmma_f16(aq1, bk1, z);
            s[kn] = z;
        }
        // ---- scale + causal mask ----
#pragma unroll
        for (int kn = 0; kn < 2; ++kn)
#pragma unroll
            for (int r = 0; r < 8; ++r) {
                const int qrow = qbase + r + g * 8;
                const int key  = j0 + kn * 16 + nl;
                const float v  = s[kn][r] * scale;
                s[kn][r] = (key <= qrow) ? v : -INFINITY;
            }
        // ---- online softmax (per-row stats replicated across 16 lanes) ----
#pragma unroll
        for (int r = 0; r < 8; ++r) {
            const float blkmax = rowmax16(fmaxf(s[0][r], s[1][r]));
            const float mnew   = fmaxf(mrow[r], blkmax);
            const float alpha  = __expf(mrow[r] - mnew);
            const float p0 = __expf(s[0][r] - mnew);
            const float p1 = __expf(s[1][r] - mnew);
            s[0][r] = p0; s[1][r] = p1;
            lrow[r] = lrow[r] * alpha + rowsum16(p0 + p1);
            mrow[r] = mnew;
#pragma unroll
            for (int dn = 0; dn < 4; ++dn) acc[dn][r] *= alpha;
        }
        // ---- P: C-layout -> A-fragment via wave-private LDS ----
#pragma unroll
        for (int kn = 0; kn < 2; ++kn)
#pragma unroll
            for (int r = 0; r < 8; ++r)
                pw[(r + g * 8) * 32 + kn * 16 + nl] = (_Float16)s[kn][r];
        asm volatile("s_wait_dscnt 0" ::: "memory");
        const v16h ap = load_a_frag(pw, 32);
        // ---- P (16x32) x V (32x64): 4 WMMA, V^T rows are contiguous ----
#pragma unroll
        for (int dn = 0; dn < 4; ++dn) {
            v16h bv = load_b_frag(Vtb + (size_t)(dn * 16) * SEQ + j0, SEQ);
            acc[dn] = wmma_f16(ap, bv, acc[dn]);
        }
    }

    // ---- normalize and store to (B, S, H*64+d) f16 ----
#pragma unroll
    for (int r = 0; r < 8; ++r) {
        const float inv = 1.0f / lrow[r];
        const int srow = qbase + r + g * 8;
#pragma unroll
        for (int dn = 0; dn < 4; ++dn) {
            O[((size_t)(b * SEQ + srow)) * D_MODEL + h * D_K + dn * 16 + nl] =
                (_Float16)(acc[dn][r] * inv);
        }
    }
}

// ---------------------------------------------------------------------------
extern "C" void kernel_launch(void* const* d_in, const int* in_sizes, int n_in,
                              void* d_out, int out_size, void* d_ws, size_t ws_size,
                              hipStream_t stream) {
    (void)in_sizes; (void)n_in; (void)out_size; (void)ws_size;
    const float* x  = (const float*)d_in[0];
    const float* Wq = (const float*)d_in[1];
    const float* Wk = (const float*)d_in[2];
    const float* Wv = (const float*)d_in[3];
    const float* Wo = (const float*)d_in[4];
    // d_in[5] is the causal mask; causality is applied analytically.

    char* ws = (char*)d_ws;
    size_t off = 0;
    const size_t act_bytes = (size_t)MTOT * D_MODEL * sizeof(_Float16);    // 16 MB
    const size_t wgt_bytes = (size_t)D_MODEL * D_MODEL * sizeof(_Float16); //  2 MB
    _Float16* x_h    = (_Float16*)(ws + off); off += act_bytes;
    _Float16* wq_h   = (_Float16*)(ws + off); off += wgt_bytes;
    _Float16* wk_h   = (_Float16*)(ws + off); off += wgt_bytes;
    _Float16* wv_h   = (_Float16*)(ws + off); off += wgt_bytes;
    _Float16* wo_h   = (_Float16*)(ws + off); off += wgt_bytes;
    _Float16* q_h    = (_Float16*)(ws + off); off += act_bytes;  // (B,H,S,64)
    _Float16* k_h    = (_Float16*)(ws + off); off += act_bytes;  // (B,H,S,64)
    _Float16* vt_h   = (_Float16*)(ws + off); off += act_bytes;  // (B,H,64,S)
    _Float16* attn_h = (_Float16*)(ws + off); off += act_bytes;  // (B,S,1024)

    // 1) f32 -> f16 conversions
    const int n4x = MTOT * D_MODEL / 4;
    const int n4w = D_MODEL * D_MODEL / 4;
    cvt_f32_to_f16_kernel<<<(n4x + 255) / 256, 256, 0, stream>>>(x,  x_h,  n4x);
    cvt_f32_to_f16_kernel<<<(n4w + 255) / 256, 256, 0, stream>>>(Wq, wq_h, n4w);
    cvt_f32_to_f16_kernel<<<(n4w + 255) / 256, 256, 0, stream>>>(Wk, wk_h, n4w);
    cvt_f32_to_f16_kernel<<<(n4w + 255) / 256, 256, 0, stream>>>(Wv, wv_h, n4w);
    cvt_f32_to_f16_kernel<<<(n4w + 255) / 256, 256, 0, stream>>>(Wo, wo_h, n4w);

    // 2) Q/K/V projections (TDM-fed WMMA GEMM); V lands pre-transposed.
    dim3 ggrid(D_MODEL / 64, MTOT / 128);   // (16, 64)
    gemm_xwT_kernel<<<ggrid, 256, 0, stream>>>(x_h, wq_h, q_h,  nullptr, 0);
    gemm_xwT_kernel<<<ggrid, 256, 0, stream>>>(x_h, wk_h, k_h,  nullptr, 0);
    gemm_xwT_kernel<<<ggrid, 256, 0, stream>>>(x_h, wv_h, vt_h, nullptr, 2);

    // 3) causal flash attention
    dim3 agrid(SEQ / 128, BATCH * N_HEADS); // (16, 64)
    attn_kernel<<<agrid, 256, 0, stream>>>(q_h, k_h, vt_h, attn_h);

    // 4) output projection -> f32 d_out
    gemm_xwT_kernel<<<ggrid, 256, 0, stream>>>(attn_h, wo_h, nullptr,
                                               (float*)d_out, 1);
}
